// GATv2Net_38697655336975
// MI455X (gfx1250) — compile-verified, matrix operand order checked by reading, not run on hardware
//
#include <hip/hip_runtime.h>
#include <hip/hip_bf16.h>

// ---------------------------------------------------------------------------
// GATv2 (2 layers) for MI455X / gfx1250.
// Fused xl|xr GEMM per layer using v_wmma_f32_16x16x32_bf16:
//   one wave = 16 rows x 4 column-tiles; per k-step: load A (2xB128) +
//   all 4 B tiles (8xB128, one clause), then 4 back-to-back WMMAs.
// Edge phase: wave-per-edge, shfl_xor reductions, f32 atomics for segment ops.
// ---------------------------------------------------------------------------

typedef __attribute__((ext_vector_type(16))) __bf16 v16bf;
typedef __attribute__((ext_vector_type(8)))  float  v8f;

#define HD 4            // heads
#define SLOPE 0.2f

static __device__ __forceinline__ unsigned fenc(float f) {
  unsigned u = __float_as_uint(f);
  return (u >> 31) ? ~u : (u | 0x80000000u);   // order-preserving float->u32
}
static __device__ __forceinline__ float fdec(unsigned k) {
  return __uint_as_float((k >> 31) ? (k & 0x7FFFFFFFu) : ~k);
}

// ---------------------------------------------------------------- utilities
__global__ void fill_u32_kernel(unsigned* __restrict__ p, unsigned v, int n) {
  int i = blockIdx.x * blockDim.x + threadIdx.x;
  if (i < n) p[i] = v;
}

__global__ void cvt_bf16_kernel(const float* __restrict__ src,
                                __bf16* __restrict__ dst, int n) {
  int i = blockIdx.x * blockDim.x + threadIdx.x;
  if (i < n) dst[i] = (__bf16)src[i];
}

__global__ void concat_bias_kernel(const float* __restrict__ a,
                                   const float* __restrict__ b,
                                   float* __restrict__ dst, int n) {
  int i = blockIdx.x * blockDim.x + threadIdx.x;
  if (i < 2 * n) dst[i] = (i < n) ? a[i] : b[i - n];
}

// Pack W [K x NC] (f32, row-major) into WMMA-B bf16 layout:
// dst[((nt*(K/32)+kt)*32 + lane)*16 + i]; lane holds col n = nt*16+(lane&15),
// half h = lane>>4; element i -> vgpr v=i/2, j=i&1 ->
// K = kt*32 + (v<4 ? 2v + 8h + j : 16 + 2(v-4) + 8h + j)   (ISA 7.12.2)
__global__ void pack_w_kernel(const float* __restrict__ W,
                              __bf16* __restrict__ dst, int K, int NC) {
  int idx = blockIdx.x * blockDim.x + threadIdx.x;
  if (idx >= K * NC) return;
  int i    = idx & 15;
  int lane = (idx >> 4) & 31;
  int t    = idx >> 9;
  int kts  = K >> 5;
  int kt   = t % kts;
  int nt   = t / kts;
  int n    = nt * 16 + (lane & 15);
  int hh   = lane >> 4;
  int v    = i >> 1, j = i & 1;
  int kk   = (v < 4) ? (v * 2 + hh * 8 + j) : (16 + (v - 4) * 2 + hh * 8 + j);
  int k    = kt * 32 + kk;
  dst[idx] = (__bf16)W[(size_t)k * NC + n];
}

// ------------------------------------------------------------------- GEMM
// Y[M x NC] = X[M x K](bf16) @ Wpacked + bias.
// One wave computes 16 rows x (TN*16) cols.
template <int K, int TN>
__global__ void gemm_wmma_kernel(const __bf16* __restrict__ X,
                                 const __bf16* __restrict__ Wp,
                                 const float* __restrict__ bias,
                                 float* __restrict__ Y,
                                 int Mtiles, int NC) {
  const int wid  = (blockIdx.x * blockDim.x + threadIdx.x) >> 5;
  const int lane = threadIdx.x & 31;
  const int ngrp = (NC >> 4) / TN;
  const int tm = wid / ngrp;
  const int tg = wid - tm * ngrp;
  if (tm >= Mtiles) return;          // wave-uniform guard: EXEC stays all-1s

  constexpr int KTS = K >> 5;
  const __bf16* arow = X + (size_t)(tm * 16 + (lane & 15)) * K + (lane >> 4) * 8;
  const __bf16* bp   = Wp + (((size_t)(tg * TN) * KTS) * 32 + lane) * 16;

  v8f c[TN] = {};
#pragma unroll
  for (int kt = 0; kt < KTS; ++kt) {
    union { uint4 u[2]; v16bf v; } ua, ub[TN];
    // issue all loads for this k-step first (one clause), then the WMMAs
    ua.u[0] = *(const uint4*)(arow + kt * 32);        // K = kt*32+8h .. +7
    ua.u[1] = *(const uint4*)(arow + kt * 32 + 16);   // K = kt*32+16+8h .. +7
#pragma unroll
    for (int t = 0; t < TN; ++t) {
      const uint4* pb = (const uint4*)(bp + ((size_t)t * KTS + kt) * 512);
      ub[t].u[0] = pb[0];
      ub[t].u[1] = pb[1];
    }
#pragma unroll
    for (int t = 0; t < TN; ++t)
      c[t] = __builtin_amdgcn_wmma_f32_16x16x32_bf16(false, ua.v, false, ub[t].v,
                                                     (short)0, c[t], false, false);
  }
  const int colbase = tg * TN * 16 + (lane & 15);
  float* yrow = Y + (size_t)(tm * 16 + (lane >> 4) * 8) * NC + colbase;
#pragma unroll
  for (int t = 0; t < TN; ++t) {
    float bv = bias[colbase + t * 16];
#pragma unroll
    for (int r = 0; r < 8; ++r)
      yrow[(size_t)r * NC + t * 16] = c[t][r] + bv;   // D: row r+8*(lane/16)
  }
}

// ------------------------------------------------------------- edge phase
// xlr holds [xl | xr] per node, row stride 2*HD*C; xr at offset HD*C.
// Wave per edge: logits e[eh] = sum_c leaky(xl[src]+xr[dst]) * att ; seg-max.
template <int C>
__global__ void edge_logits_kernel(const int* __restrict__ ei, int E, int E2,
                                   const float* __restrict__ xlr,
                                   const float* __restrict__ att,
                                   float* __restrict__ elog,
                                   unsigned* __restrict__ menc) {
  constexpr int STR = 2 * HD * C;
  const int w = (blockIdx.x * blockDim.x + threadIdx.x) >> 5;
  const int lane = threadIdx.x & 31;
  if (w >= E2) return;
  int s, d;
  if (w < E) { s = ei[w]; d = ei[E + w]; } else { s = d = w - E; }
  const float* xls = xlr + (size_t)s * STR;            // xl half
  const float* xrd = xlr + (size_t)d * STR + HD * C;   // xr half
#pragma unroll
  for (int h = 0; h < HD; ++h) {
    float g = xls[h * C + lane] + xrd[h * C + lane];
    g = g > 0.f ? g : SLOPE * g;
    float v = g * att[h * C + lane];
    if (C > 32 && lane < C - 32) {
      int c2 = 32 + lane;
      float g2 = xls[h * C + c2] + xrd[h * C + c2];
      g2 = g2 > 0.f ? g2 : SLOPE * g2;
      v += g2 * att[h * C + c2];
    }
#pragma unroll
    for (int m = 16; m >= 1; m >>= 1) v += __shfl_xor(v, m, 32);
    if (lane == 0) {
      elog[(size_t)w * HD + h] = v;
      atomicMax(&menc[(size_t)d * HD + h], fenc(v));
    }
  }
}

// a = exp(e - m[dst]);  s[dst] += a;  acc[dst] += a * xl[src]
template <int C>
__global__ void edge_accum_kernel(const int* __restrict__ ei, int E, int E2,
                                  const float* __restrict__ xlr,
                                  const float* __restrict__ elog,
                                  const unsigned* __restrict__ menc,
                                  float* __restrict__ ssum,
                                  float* __restrict__ acc) {
  constexpr int STR = 2 * HD * C;
  const int w = (blockIdx.x * blockDim.x + threadIdx.x) >> 5;
  const int lane = threadIdx.x & 31;
  if (w >= E2) return;
  int s, d;
  if (w < E) { s = ei[w]; d = ei[E + w]; } else { s = d = w - E; }
  const float* xls = xlr + (size_t)s * STR;            // xl half
  float* accd = acc + (size_t)d * (HD * C);
#pragma unroll
  for (int h = 0; h < HD; ++h) {
    float a = __expf(elog[(size_t)w * HD + h] - fdec(menc[(size_t)d * HD + h]));
    if (lane == 0) atomicAdd(&ssum[(size_t)d * HD + h], a);
    atomicAdd(&accd[h * C + lane], a * xls[h * C + lane]);
    if (C > 32 && lane < C - 32)
      atomicAdd(&accd[h * C + 32 + lane], a * xls[h * C + 32 + lane]);
  }
}

// layer-1 epilogue: h = ELU(acc/s + bias1) -> bf16 for layer-2 GEMM
__global__ void finish1_kernel(const float* __restrict__ acc,
                               const float* __restrict__ ssum,
                               const float* __restrict__ bias,
                               __bf16* __restrict__ hbf, int N) {
  int idx = blockIdx.x * blockDim.x + threadIdx.x;
  if (idx >= N * 128) return;
  int n = idx >> 7, j = idx & 127, h = j >> 5;
  float v = acc[idx] / (ssum[n * 4 + h] + 1e-16f) + bias[j];
  v = v > 0.f ? v : expm1f(v);
  hbf[idx] = (__bf16)v;
}

// layer-2 epilogue: out = mean_h(acc/s) + bias2
__global__ void finish2_kernel(const float* __restrict__ acc,
                               const float* __restrict__ ssum,
                               const float* __restrict__ bias,
                               float* __restrict__ out, int N) {
  int idx = blockIdx.x * blockDim.x + threadIdx.x;
  if (idx >= N * 40) return;
  int n = idx / 40, o = idx - n * 40;
  float v = 0.f;
#pragma unroll
  for (int h = 0; h < 4; ++h)
    v += acc[(size_t)n * 160 + h * 40 + o] / (ssum[n * 4 + h] + 1e-16f);
  out[idx] = 0.25f * v + bias[o];
}

// -------------------------------------------------------------------- host
static inline int cdiv(long long a, long long b) { return (int)((a + b - 1) / b); }

extern "C" void kernel_launch(void* const* d_in, const int* in_sizes, int n_in,
                              void* d_out, int out_size, void* d_ws, size_t ws_size,
                              hipStream_t stream) {
  constexpr int N = 50000, E = 800000, IN = 256;
  constexpr int E2 = E + N;                 // edges incl. self loops
  const float* x     = (const float*)d_in[0];
  const int*   ei    = (const int*)d_in[1];
  const float* Wl1   = (const float*)d_in[2];
  const float* bl1   = (const float*)d_in[3];
  const float* Wr1   = (const float*)d_in[4];
  const float* br1   = (const float*)d_in[5];
  const float* att1  = (const float*)d_in[6];
  const float* bias1 = (const float*)d_in[7];
  const float* Wl2   = (const float*)d_in[8];
  const float* bl2   = (const float*)d_in[9];
  const float* Wr2   = (const float*)d_in[10];
  const float* br2   = (const float*)d_in[11];
  const float* att2  = (const float*)d_in[12];
  const float* bias2 = (const float*)d_in[13];
  float* out = (float*)d_out;

  // workspace layout (256B aligned)
  char* base = (char*)d_ws;
  size_t off = 0;
  auto take = [&](size_t bytes) -> char* {
    char* p = base + off;
    off = (off + bytes + 255) & ~(size_t)255;
    return p;
  };
  __bf16*   xbf   = (__bf16*)take((size_t)N * IN * 2);
  __bf16*   wp1   = (__bf16*)take((size_t)256 * 256 * 2);   // [Wl1|Wr1] packed
  __bf16*   wp2   = (__bf16*)take((size_t)128 * 320 * 2);   // [Wl2|Wr2] packed
  float*    cb1   = (float*)take(256 * 4);                  // bl1|br1
  float*    cb2   = (float*)take(320 * 4);                  // bl2|br2
  float*    xlr1  = (float*)take((size_t)N * 256 * 4);      // [xl1|xr1]
  float*    elog  = (float*)take((size_t)E2 * 4 * 4);
  unsigned* menc  = (unsigned*)take((size_t)N * 4 * 4);
  float*    ssum  = (float*)take((size_t)N * 4 * 4);
  float*    acc1  = (float*)take((size_t)N * 128 * 4);
  __bf16*   hbf   = (__bf16*)take((size_t)N * 128 * 2);
  float*    xlr2  = (float*)take((size_t)N * 320 * 4);      // [xl2|xr2]
  float*    acc2  = (float*)take((size_t)N * 160 * 4);
  (void)ws_size; (void)in_sizes; (void)n_in; (void)out_size;

  const unsigned ENC_NEG_INF = 0x007FFFFFu;   // fenc(-inf)
  const int Mtiles = N / 16;                  // 3125 exactly

  // x -> bf16; pack [Wl|Wr] per layer (Wr tiles start at element K*NCl);
  // concat biases.
  cvt_bf16_kernel<<<cdiv((long long)N * IN, 256), 256, 0, stream>>>(x, xbf, N * IN);
  pack_w_kernel<<<cdiv(256 * 128, 256), 256, 0, stream>>>(Wl1, wp1, 256, 128);
  pack_w_kernel<<<cdiv(256 * 128, 256), 256, 0, stream>>>(Wr1, wp1 + (size_t)256 * 128, 256, 128);
  pack_w_kernel<<<cdiv(128 * 160, 256), 256, 0, stream>>>(Wl2, wp2, 128, 160);
  pack_w_kernel<<<cdiv(128 * 160, 256), 256, 0, stream>>>(Wr2, wp2 + (size_t)128 * 160, 128, 160);
  concat_bias_kernel<<<1, 256, 0, stream>>>(bl1, br1, cb1, 128);
  concat_bias_kernel<<<2, 256, 0, stream>>>(bl2, br2, cb2, 160);

  // ---- layer 1 fused GEMM: [N,256] @ [256,256] -> xlr1
  {
    int waves = Mtiles * ((256 / 16) / 4);
    int blocks = cdiv((long long)waves * 32, 128);
    gemm_wmma_kernel<256, 4><<<blocks, 128, 0, stream>>>(xbf, wp1, cb1, xlr1, Mtiles, 256);
  }
  // segment state init
  fill_u32_kernel<<<cdiv(N * 4, 256), 256, 0, stream>>>(menc, ENC_NEG_INF, N * 4);
  fill_u32_kernel<<<cdiv(N * 4, 256), 256, 0, stream>>>((unsigned*)ssum, 0u, N * 4);
  fill_u32_kernel<<<cdiv(N * 128, 256), 256, 0, stream>>>((unsigned*)acc1, 0u, N * 128);
  // edge phase (C=32)
  {
    int blocks = cdiv((long long)E2 * 32, 256);
    edge_logits_kernel<32><<<blocks, 256, 0, stream>>>(ei, E, E2, xlr1, att1, elog, menc);
    edge_accum_kernel<32><<<blocks, 256, 0, stream>>>(ei, E, E2, xlr1, elog, menc, ssum, acc1);
  }
  finish1_kernel<<<cdiv(N * 128, 256), 256, 0, stream>>>(acc1, ssum, bias1, hbf, N);

  // ---- layer 2 fused GEMM: [N,128] @ [128,320] -> xlr2
  {
    int waves = Mtiles * ((320 / 16) / 4);
    int blocks = cdiv((long long)waves * 32, 128);
    gemm_wmma_kernel<128, 4><<<blocks, 128, 0, stream>>>(hbf, wp2, cb2, xlr2, Mtiles, 320);
  }
  fill_u32_kernel<<<cdiv(N * 4, 256), 256, 0, stream>>>(menc, ENC_NEG_INF, N * 4);
  fill_u32_kernel<<<cdiv(N * 4, 256), 256, 0, stream>>>((unsigned*)ssum, 0u, N * 4);
  fill_u32_kernel<<<cdiv(N * 160, 256), 256, 0, stream>>>((unsigned*)acc2, 0u, N * 160);
  // edge phase (C=40)
  {
    int blocks = cdiv((long long)E2 * 32, 256);
    edge_logits_kernel<40><<<blocks, 256, 0, stream>>>(ei, E, E2, xlr2, att2, elog, menc);
    edge_accum_kernel<40><<<blocks, 256, 0, stream>>>(ei, E, E2, xlr2, elog, menc, ssum, acc2);
  }
  finish2_kernel<<<cdiv(N * 40, 256), 256, 0, stream>>>(acc2, ssum, bias2, out, N);
}
